// EncoderLayer_10651518894949
// MI455X (gfx1250) — matrix operand order repaired and checked by reading, not compiled, not run on hardware
//
#include <hip/hip_runtime.h>
#include <stdint.h>

#define DM 512
#define DFF 2048
#define DK 64
#define NH 8

typedef unsigned short bf16_t;
typedef __attribute__((ext_vector_type(16))) __bf16 bfrag;
typedef __attribute__((ext_vector_type(8)))  float  v8f;
typedef __attribute__((ext_vector_type(4)))  unsigned int u32x4;
typedef __attribute__((ext_vector_type(4)))  int i32x4;

using gvec_t = __attribute__((address_space(1))) i32x4*;
using lvec_t = __attribute__((address_space(3))) i32x4*;

__device__ inline bf16_t f2bf(float f) {
  union { float f; unsigned u; } c; c.f = f;
  unsigned r = c.u + 0x7FFFu + ((c.u >> 16) & 1u);   // round-to-nearest-even
  return (bf16_t)(r >> 16);
}

__device__ inline v8f v8zero() {
  v8f z = {0.f, 0.f, 0.f, 0.f, 0.f, 0.f, 0.f, 0.f};
  return z;
}

// Load a 32-byte WMMA fragment as two 16-byte chunks (p0, p1).
__device__ inline bfrag ld_frag2(const bf16_t* p0, const bf16_t* p1) {
  union { bfrag f; u32x4 q[2]; } u;
  u.q[0] = *(const u32x4*)p0;
  u.q[1] = *(const u32x4*)p1;
  return u.f;
}

__device__ inline v8f wmma_bf16(bfrag a, bfrag b, v8f c) {
  return __builtin_amdgcn_wmma_f32_16x16x32_bf16(false, a, false, b,
                                                 (short)0, c, false, false);
}

// --- async global->LDS copy (16B per call), verified to lower to
// global_load_async_to_lds_b128 on this toolchain ---
__device__ inline void cp_async16(void* l, const void* g) {
#if __has_builtin(__builtin_amdgcn_global_load_async_to_lds_b128)
  __builtin_amdgcn_global_load_async_to_lds_b128(
      (gvec_t)(uintptr_t)g, (lvec_t)(unsigned int)(uintptr_t)l, 0, 0);
#else
  *(u32x4*)l = *(const u32x4*)g;
#endif
}

__device__ inline void wait_async0() {
#if __has_builtin(__builtin_amdgcn_s_wait_asynccnt)
  __builtin_amdgcn_s_wait_asynccnt(0);
#else
  asm volatile("s_wait_asynccnt 0" ::: "memory");
#endif
}

// --- shuffle reductions ---
__device__ inline float redmax16(float v) {
#pragma unroll
  for (int o = 1; o < 16; o <<= 1) v = fmaxf(v, __shfl_xor(v, o, 16));
  return v;
}
__device__ inline float redsum16(float v) {
#pragma unroll
  for (int o = 1; o < 16; o <<= 1) v += __shfl_xor(v, o, 16);
  return v;
}
__device__ inline float redsum32(float v) {
#pragma unroll
  for (int o = 1; o < 32; o <<= 1) v += __shfl_xor(v, o, 32);
  return v;
}

// ---------------------------------------------------------------------------
// fp32 -> bf16 elementwise
// ---------------------------------------------------------------------------
__global__ void cvt_bf16_kernel(const float* __restrict__ s,
                                bf16_t* __restrict__ d, int n) {
  int i = blockIdx.x * 256 + threadIdx.x;
  if (i < n) d[i] = f2bf(s[i]);
}

// ---------------------------------------------------------------------------
// C[M,N] = A[M,K] * W[N,K]^T + bias[N], optional ReLU, fp32 and/or bf16 out.
// 256 threads = 8 waves; block tile 256x128; K-step 32; double-buffered LDS
// staged with global_load_async_to_lds_b128.
// Wave grid 4(M) x 2(N); each wave computes 64x64 = 4x4 tiles of 16x16:
// 16 v_wmma_f32_16x16x32_bf16 per K-step fed by 8 LDS fragment loads.
// ---------------------------------------------------------------------------
__global__ __launch_bounds__(256) void
gemm_bf16_wmma(const bf16_t* __restrict__ A, const bf16_t* __restrict__ W,
               const float* __restrict__ bias, float* __restrict__ Cf,
               bf16_t* __restrict__ Cb, int M, int N, int K, int relu) {
  (void)M;
  __shared__ __align__(16) bf16_t As[2][256 * 32];   // 16KB x2
  __shared__ __align__(16) bf16_t Ws[2][128 * 32];   // 8KB x2

  const int tid  = threadIdx.x;
  const int lane = tid & 31;
  const int wave = tid >> 5;
  const int bm = blockIdx.y * 256;
  const int bn = blockIdx.x * 128;

  const int wm = (wave >> 1) << 6;    // wave M offset: 0,64,128,192
  const int wn = (wave & 1) << 6;     // wave N offset: 0,64
  const int am = lane & 15;
  const int ak = (lane >> 4) << 3;    // A-frag K sub-offset (0 or 8)
  const int bk = (lane >> 4) << 3;    // B-frag K sub-offset (0 or 8)

  v8f acc[4][4];
#pragma unroll
  for (int i = 0; i < 4; ++i)
#pragma unroll
    for (int j = 0; j < 4; ++j) acc[i][j] = v8zero();

  // staging map: A -> thread t stages row t (32 elems = 4x16B);
  //              W -> thread t stages row t/2, half (t&1) (2x16B)
  const int wr = tid >> 1;
  const int wc = (tid & 1) << 4;

  auto load_tiles = [&](int buf, int k0) {
    const bf16_t* ga = A + (size_t)(bm + tid) * K + k0;
    bf16_t* la = &As[buf][tid * 32];
#pragma unroll
    for (int c = 0; c < 4; ++c) cp_async16(la + c * 8, ga + c * 8);
    const bf16_t* gw = W + (size_t)(bn + wr) * K + k0 + wc;
    bf16_t* lw = &Ws[buf][wr * 32 + wc];
#pragma unroll
    for (int c = 0; c < 2; ++c) cp_async16(lw + c * 8, gw + c * 8);
  };

  const int nk = K >> 5;
  load_tiles(0, 0);
  wait_async0();
  __syncthreads();

  for (int kb = 0; kb < nk; ++kb) {
    const int buf = kb & 1;
    if (kb + 1 < nk) load_tiles(buf ^ 1, (kb + 1) << 5);

    bfrag af[4], wf[4];
#pragma unroll
    for (int i = 0; i < 4; ++i) {
      // A 16x32 bf16 layout: lane<16 -> K {ak..ak+7, ak+16..ak+23}
      const bf16_t* p = &As[buf][(wm + i * 16 + am) * 32 + ak];
      af[i] = ld_frag2(p, p + 16);
    }
#pragma unroll
    for (int j = 0; j < 4; ++j) {
      // B 32x16 bf16 layout: lane holds col=am, K {bk..bk+7, bk+16..bk+23}
      const bf16_t* p = &Ws[buf][(wn + j * 16 + am) * 32 + bk];
      wf[j] = ld_frag2(p, p + 16);
    }
#pragma unroll
    for (int i = 0; i < 4; ++i)
#pragma unroll
      for (int j = 0; j < 4; ++j) acc[i][j] = wmma_bf16(af[i], wf[j], acc[i][j]);

    wait_async0();
    __syncthreads();
  }

  const int rowoff = (lane >> 4) << 3;   // C tile: vgpr e -> row e (+8 hi half)
#pragma unroll
  for (int i = 0; i < 4; ++i)
#pragma unroll
    for (int j = 0; j < 4; ++j) {
      const int c = bn + wn + j * 16 + am;
      const float bv = bias[c];
#pragma unroll
      for (int e = 0; e < 8; ++e) {
        const int r = bm + wm + i * 16 + rowoff + e;
        float v = acc[i][j][e] + bv;
        if (relu) v = fmaxf(v, 0.f);
        if (Cf) Cf[(size_t)r * N + c] = v;
        if (Cb) Cb[(size_t)r * N + c] = f2bf(v);
      }
    }
}

// ---------------------------------------------------------------------------
// Flash attention with Q=K=V = qkv (bf16). One block per (b,h,128 q-rows),
// 8 waves x 16 q-rows. K/V staged in LDS in both [key][d] and [d][key] forms.
// ---------------------------------------------------------------------------
__global__ __launch_bounds__(256) void
flash_attn_wmma(const bf16_t* __restrict__ Hq, bf16_t* __restrict__ Ctx) {
  __shared__ __align__(16) bf16_t Ks[32 * 64];     // [key][d]
  __shared__ __align__(16) bf16_t Kt[64 * 32];     // [d][key] (transposed)
  __shared__ __align__(16) bf16_t Ps[8][16 * 32];  // per-wave P staging

  const int S = 1024;
  const int tid = threadIdx.x, lane = tid & 31, wave = tid >> 5;
  const int qblk = blockIdx.x & 7;          // S/128 = 8
  const int h = (blockIdx.x >> 3) & 7;
  const int b = blockIdx.x >> 6;
  const size_t base = (size_t)b * S * DM;
  const int hd0 = h * DK;
  const int q0 = qblk * 128 + wave * 16;
  const int am = lane & 15;
  const int ak = (lane >> 4) << 3;
  const int bk = (lane >> 4) << 3;

  // Q fragments: 16x64 split into two 16x32 A-fragments (loaded from global)
  bfrag aq[2];
#pragma unroll
  for (int d = 0; d < 2; ++d) {
    const bf16_t* p = Hq + base + (size_t)(q0 + am) * DM + hd0 + d * 32 + ak;
    aq[d] = ld_frag2(p, p + 16);
  }

  v8f accc[4];
#pragma unroll
  for (int j = 0; j < 4; ++j) accc[j] = v8zero();
  v8f mrun, lsum;
#pragma unroll
  for (int e = 0; e < 8; ++e) { mrun[e] = -1e30f; lsum[e] = 0.f; }

  const int sr = tid >> 3;           // staging: key row 0..31
  const int scc = (tid & 7) << 3;    // staging: d col 0..56 step 8

  for (int kb = 0; kb < S / 32; ++kb) {
    const int key0 = kb * 32;
    __syncthreads();   // protect LDS tiles from previous iteration readers
    {
      const bf16_t* g = Hq + base + (size_t)(key0 + sr) * DM + hd0 + scc;
      u32x4 v = *(const u32x4*)g;
      *(u32x4*)&Ks[sr * 64 + scc] = v;
      const bf16_t* pv = (const bf16_t*)&v;
#pragma unroll
      for (int t = 0; t < 8; ++t) Kt[(scc + t) * 32 + sr] = pv[t];
    }
    __syncthreads();

    // scores: two 16x16 tiles (keys 0-15 and 16-31), K-dim = d (64 = 2x32)
    v8f sc0 = v8zero(), sc1 = v8zero();
#pragma unroll
    for (int d = 0; d < 2; ++d) {
      const bf16_t* p0 = &Ks[am * 64 + d * 32 + bk];
      const bf16_t* p1 = &Ks[(16 + am) * 64 + d * 32 + bk];
      sc0 = wmma_bf16(aq[d], ld_frag2(p0, p0 + 16), sc0);
      sc1 = wmma_bf16(aq[d], ld_frag2(p1, p1 + 16), sc1);
    }

    // online softmax: rows live across 16 lanes of each half-wave
    v8f mnew, p0v, p1v, alpha;
#pragma unroll
    for (int e = 0; e < 8; ++e) {
      float s0 = sc0[e] * 0.125f;            // 1/sqrt(64)
      float s1 = sc1[e] * 0.125f;
      float mx = redmax16(fmaxf(s0, s1));
      mnew[e]  = fmaxf(mrun[e], mx);
      alpha[e] = __expf(mrun[e] - mnew[e]);
      p0v[e]   = __expf(s0 - mnew[e]);
      p1v[e]   = __expf(s1 - mnew[e]);
      float rs = redsum16(p0v[e] + p1v[e]);
      lsum[e]  = lsum[e] * alpha[e] + rs;
      mrun[e]  = mnew[e];
    }
#pragma unroll
    for (int j = 0; j < 4; ++j)
#pragma unroll
      for (int e = 0; e < 8; ++e) accc[j][e] *= alpha[e];

    // repack P (C-layout) -> A-fragment via per-wave LDS scratch
    {
      const int rowoff = (lane >> 4) << 3;
#pragma unroll
      for (int e = 0; e < 8; ++e) {
        Ps[wave][(rowoff + e) * 32 + am]      = f2bf(p0v[e]);
        Ps[wave][(rowoff + e) * 32 + 16 + am] = f2bf(p1v[e]);
      }
    }
    const bf16_t* pp = &Ps[wave][am * 32 + ak];
    bfrag ap = ld_frag2(pp, pp + 16);

    // ctx += P(16x32) @ V(32x64): B-fragments from transposed tile Kt[d][key]
#pragma unroll
    for (int j = 0; j < 4; ++j) {
      const bf16_t* p = &Kt[(j * 16 + am) * 32 + bk];
      accc[j] = wmma_bf16(ap, ld_frag2(p, p + 16), accc[j]);
    }
  }

  const int rowoff = (lane >> 4) << 3;
#pragma unroll
  for (int j = 0; j < 4; ++j)
#pragma unroll
    for (int e = 0; e < 8; ++e) {
      const int m = rowoff + e;
      Ctx[(size_t)(b * S + q0 + m) * DM + hd0 + j * 16 + am] =
          f2bf(accc[j][e] / lsum[e]);
    }
}

// ---------------------------------------------------------------------------
// x1 = LN(x)*g+be + LN(attn)*g+be = (xhat + ahat)*g + 2*be   (wave per row)
// ---------------------------------------------------------------------------
__global__ __launch_bounds__(256) void
ln_dual_kernel(const float* __restrict__ X, const float* __restrict__ Atn,
               const float* __restrict__ g, const float* __restrict__ be,
               float* __restrict__ Out, bf16_t* __restrict__ OutB) {
  const int lane = threadIdx.x & 31, wave = threadIdx.x >> 5;
  const size_t row = (size_t)blockIdx.x * 8 + wave;
  const float* x = X + row * DM;
  const float* a = Atn + row * DM;
  float xv[16], av[16];
  float xs = 0.f, xs2 = 0.f, as = 0.f, as2 = 0.f;
#pragma unroll
  for (int i = 0; i < 16; ++i) {
    float v = x[lane + i * 32]; xv[i] = v; xs += v; xs2 += v * v;
    float w = a[lane + i * 32]; av[i] = w; as += w; as2 += w * w;
  }
  xs = redsum32(xs); xs2 = redsum32(xs2);
  as = redsum32(as); as2 = redsum32(as2);
  const float inv = 1.f / DM;
  const float xmu = xs * inv, amu = as * inv;
  const float xr = rsqrtf(fmaxf(xs2 * inv - xmu * xmu, 0.f) + 1e-5f);
  const float ar = rsqrtf(fmaxf(as2 * inv - amu * amu, 0.f) + 1e-5f);
#pragma unroll
  for (int i = 0; i < 16; ++i) {
    const int c = lane + i * 32;
    const float v = ((xv[i] - xmu) * xr + (av[i] - amu) * ar) * g[c] + 2.f * be[c];
    Out[row * DM + c] = v;
    OutB[row * DM + c] = f2bf(v);
  }
}

// ---------------------------------------------------------------------------
// out = x1 + LN(ffn)*g + be
// ---------------------------------------------------------------------------
__global__ __launch_bounds__(256) void
ln_final_kernel(const float* __restrict__ X1, const float* __restrict__ F,
                const float* __restrict__ g, const float* __restrict__ be,
                float* __restrict__ Out) {
  const int lane = threadIdx.x & 31, wave = threadIdx.x >> 5;
  const size_t row = (size_t)blockIdx.x * 8 + wave;
  const float* f = F + row * DM;
  float fv[16];
  float s = 0.f, s2 = 0.f;
#pragma unroll
  for (int i = 0; i < 16; ++i) {
    float v = f[lane + i * 32]; fv[i] = v; s += v; s2 += v * v;
  }
  s = redsum32(s); s2 = redsum32(s2);
  const float inv = 1.f / DM;
  const float mu = s * inv;
  const float r = rsqrtf(fmaxf(s2 * inv - mu * mu, 0.f) + 1e-5f);
#pragma unroll
  for (int i = 0; i < 16; ++i) {
    const int c = lane + i * 32;
    Out[row * DM + c] = X1[row * DM + c] + (fv[i] - mu) * r * g[c] + be[c];
  }
}

// ---------------------------------------------------------------------------
extern "C" void kernel_launch(void* const* d_in, const int* in_sizes, int n_in,
                              void* d_out, int out_size, void* d_ws,
                              size_t ws_size, hipStream_t stream) {
  (void)n_in; (void)out_size; (void)ws_size;
  const float* xf   = (const float*)d_in[0];
  const float* wqkv = (const float*)d_in[1];
  const float* bqkv = (const float*)d_in[2];
  const float* wout = (const float*)d_in[3];
  const float* bout = (const float*)d_in[4];
  const float* w1   = (const float*)d_in[5];
  const float* b1   = (const float*)d_in[6];
  const float* w2   = (const float*)d_in[7];
  const float* b2   = (const float*)d_in[8];
  const float* g1   = (const float*)d_in[9];
  const float* be1  = (const float*)d_in[10];
  const float* g2   = (const float*)d_in[11];
  const float* be2  = (const float*)d_in[12];

  const int Mrows = in_sizes[0] / DM;   // B*S = 16384

  char* wp = (char*)d_ws;
  auto alloc = [&](size_t bytes) -> void* {
    void* p = (void*)wp;
    wp += (bytes + 255) & ~(size_t)255;
    return p;
  };
  bf16_t* xb    = (bf16_t*)alloc((size_t)Mrows * DM * 2);
  bf16_t* wqkvb = (bf16_t*)alloc((size_t)DM * DM * 2);
  bf16_t* woutb = (bf16_t*)alloc((size_t)DM * DM * 2);
  bf16_t* w1b   = (bf16_t*)alloc((size_t)DFF * DM * 2);
  bf16_t* w2b   = (bf16_t*)alloc((size_t)DM * DFF * 2);
  bf16_t* qkvb  = (bf16_t*)alloc((size_t)Mrows * DM * 2);
  bf16_t* ctxb  = (bf16_t*)alloc((size_t)Mrows * DM * 2);
  float*  attn  = (float*)alloc((size_t)Mrows * DM * 4);
  float*  x1    = (float*)alloc((size_t)Mrows * DM * 4);
  bf16_t* x1b   = (bf16_t*)alloc((size_t)Mrows * DM * 2);
  bf16_t* y1b   = (bf16_t*)alloc((size_t)Mrows * DFF * 2);
  float*  ffn   = (float*)alloc((size_t)Mrows * DM * 4);

  auto cvt = [&](const float* s, bf16_t* d, size_t n) {
    cvt_bf16_kernel<<<dim3((unsigned)((n + 255) / 256)), dim3(256), 0, stream>>>(
        s, d, (int)n);
  };
  cvt(xf, xb, (size_t)Mrows * DM);
  cvt(wqkv, wqkvb, (size_t)DM * DM);
  cvt(wout, woutb, (size_t)DM * DM);
  cvt(w1, w1b, (size_t)DFF * DM);
  cvt(w2, w2b, (size_t)DM * DFF);

  const dim3 blk(256);
  // 1) shared QKV projection -> bf16
  gemm_bf16_wmma<<<dim3(DM / 128, Mrows / 256), blk, 0, stream>>>(
      xb, wqkvb, bqkv, nullptr, qkvb, Mrows, DM, DM, 0);
  // 2) flash attention (Q=K=V) -> bf16 ctx
  flash_attn_wmma<<<dim3((unsigned)((Mrows / 128) * NH)), blk, 0, stream>>>(
      qkvb, ctxb);
  // 3) output projection -> fp32 attn
  gemm_bf16_wmma<<<dim3(DM / 128, Mrows / 256), blk, 0, stream>>>(
      ctxb, woutb, bout, attn, nullptr, Mrows, DM, DM, 0);
  // 4) x1 = LN(x) + LN(attn)
  ln_dual_kernel<<<dim3(Mrows / 8), blk, 0, stream>>>(xf, attn, g1, be1, x1, x1b);
  // 5) FFN layer 1 (+ReLU) -> bf16 hidden
  gemm_bf16_wmma<<<dim3(DFF / 128, Mrows / 256), blk, 0, stream>>>(
      x1b, w1b, b1, nullptr, y1b, Mrows, DFF, DM, 1);
  // 6) FFN layer 2 -> fp32
  gemm_bf16_wmma<<<dim3(DM / 128, Mrows / 256), blk, 0, stream>>>(
      y1b, w2b, b2, ffn, nullptr, Mrows, DM, DFF, 0);
  // 7) out = x1 + LN(ffn)
  ln_final_kernel<<<dim3(Mrows / 8), blk, 0, stream>>>(x1, ffn, g2, be2,
                                                       (float*)d_out);
}
